// VideoEncoderMemory_82841329205709
// MI455X (gfx1250) — matrix operand order repaired and checked by reading, not compiled, not run on hardware
//
#include <hip/hip_runtime.h>

// ---------------------------------------------------------------------------
// Types for gfx1250 WMMA (wave32)
// ---------------------------------------------------------------------------
typedef __bf16 bf16_t;
typedef __attribute__((ext_vector_type(16))) __bf16 v16bf;
typedef __attribute__((ext_vector_type(8)))  __bf16 v8bf;
typedef __attribute__((ext_vector_type(4)))  __bf16 v4bf;
typedef __attribute__((ext_vector_type(8)))  float  v8f;
typedef __attribute__((ext_vector_type(4)))  float  v4f;

#define DEV static __device__ __forceinline__

constexpr int B  = 32;
constexpr int T  = 64;
constexpr int H  = 1024;
constexpr int H2 = 2 * H;
constexpr int H4 = 4 * H;
constexpr int BT = B * T;

// ---------------------------------------------------------------------------
// WMMA operand loaders (contiguous k = 16*hl + s in both A and B, 16-bit A 16x32 / B 32x16)
// ---------------------------------------------------------------------------
// A (16x32 bf16, row-major source, lane half 'hl'):
//   elems0..15 = K[k+hl*16 .. k+hl*16+15] (contiguous, matches B)
DEV v16bf load_a_bf(const bf16_t* __restrict__ row, int k, int hl) {
  return *(const v16bf*)(row + k + hl * 16);
}

// B (32x16 bf16) where B = W^T and W is row-major (N rows, K cols):
// lane's column n holds K[k+hl*16 .. k+hl*16+15] -> one contiguous 32B load.
DEV v16bf load_b_bf(const bf16_t* __restrict__ wrow, int k, int hl) {
  return *(const v16bf*)(wrow + k + hl * 16);
}

DEV v8f wmma_bf16(v16bf a, v16bf b, v8f c) {
  return __builtin_amdgcn_wmma_f32_16x16x32_bf16(false, a, false, b,
                                                 (short)0, c, false, false);
}

DEV float sigmoidf_(float x) { return 1.0f / (1.0f + __expf(-x)); }

// ---------------------------------------------------------------------------
// Kernel: fp32 -> bf16 conversion (weights + inputs converted once per launch;
// bf16 weights (~92 MB) stay resident in the 192 MB L2 across the recurrence)
// ---------------------------------------------------------------------------
__global__ void k_f32_to_bf16(const float* __restrict__ src,
                              bf16_t* __restrict__ dst, int n4) {
  int i      = blockIdx.x * blockDim.x + threadIdx.x;
  int stride = gridDim.x * blockDim.x;
  for (; i < n4; i += stride) {
    v4f v = *(const v4f*)(src + (size_t)i * 4);
    v4bf o;
#pragma unroll
    for (int j = 0; j < 4; ++j) o[j] = (__bf16)v[j];
    *(v4bf*)(dst + (size_t)i * 4) = o;
  }
}

// Zero-fill (h/c state region) — deterministic re-init every launch.
__global__ void k_zero_u64(unsigned long long* __restrict__ p, int n) {
  int i      = blockIdx.x * blockDim.x + threadIdx.x;
  int stride = gridDim.x * blockDim.x;
  for (; i < n; i += stride) p[i] = 0ull;
}

// ---------------------------------------------------------------------------
// Kernel: precompute Gx[t*B+b][j] = x[b][t] @ Wih^T + bih + bhh   (parallel)
// M = T*B = 2048 (128 tiles), N = 4H = 4096 (64 groups of 4 tiles), K = 2048.
// Register blocking: each wave owns 1 M-tile x 4 N-tiles -> one A load feeds
// four WMMA accumulator chains (4x less A traffic from L2).
// ---------------------------------------------------------------------------
__global__ void k_gemm_gx(const bf16_t* __restrict__ X,  // (B,T,2H) bf16
                          const bf16_t* __restrict__ W,  // (4H,2H) bf16
                          const float* __restrict__ bih,
                          const float* __restrict__ bhh,
                          float* __restrict__ Gx)        // (T*B,4H) fp32
{
  const int lane = threadIdx.x & 31;
  const int hl   = lane >> 4;
  const int l    = lane & 15;
  const int wid  = blockIdx.x * (blockDim.x >> 5) + (threadIdx.x >> 5);
  const int ng   = wid & 63;    // 64 groups of 4 N tiles
  const int mt   = wid >> 6;    // 128 M tiles

  const int arow = mt * 16 + l;               // row index r = t*B + b
  const int b    = arow & (B - 1);
  const int t    = arow >> 5;
  const bf16_t* xrow = X + (size_t)(b * T + t) * H2;
  const bf16_t* wrow[4];
#pragma unroll
  for (int j = 0; j < 4; ++j)
    wrow[j] = W + (size_t)(ng * 64 + j * 16 + l) * H2;

  v8f acc[4] = {v8f{}, v8f{}, v8f{}, v8f{}};
  for (int k = 0; k < H2; k += 32) {
    v16bf a = load_a_bf(xrow, k, hl);
    __builtin_prefetch(wrow[0] + k + 256, 0, 1);   // stream B rows ahead
#pragma unroll
    for (int j = 0; j < 4; ++j)
      acc[j] = wmma_bf16(a, load_b_bf(wrow[j], k, hl), acc[j]);
  }

#pragma unroll
  for (int j = 0; j < 4; ++j) {
    const int   n    = ng * 64 + j * 16 + l;
    const float bias = bih[n] + bhh[n];
#pragma unroll
    for (int r = 0; r < 8; ++r) {
      const int m = mt * 16 + r + hl * 8;
      Gx[(size_t)m * H4 + n] = acc[j][r] + bias;
    }
  }
}

// ---------------------------------------------------------------------------
// Dual LSTM cell step: blockIdx.y selects appearance/motion parameter set so
// the two independent streams share one launch (halves the serial critical
// path: 128 dependent launches instead of 256).
// gates(32 x 4H) = base + A1(32xH) @ W1^T [+ A2(32xH) @ W2^T]
// Each wave owns one hidden-column group and computes all four gate quadrants
// (i,f,g,o) as 4 WMMA accumulator chains sharing operand A.
// ---------------------------------------------------------------------------
struct StepArgs {
  const bf16_t* A1;     // (B,H) bf16 recurrent/stacked input
  const bf16_t* W1;     // (4H,H) bf16
  const bf16_t* A2;     // nullable second operand (cell-2 h_prev)
  const bf16_t* W2;     // (4H,H) bf16
  const float*  Gbase;  // (B,4H) precomputed x-part+biases, or null
  const float*  bih;    // used iff !Gbase
  const float*  bhh;
  float*        c;      // (B,H) fp32 state, updated in place
  bf16_t*       h_out;  // (B,H) bf16 (ping-pong slot)
  bf16_t*       h_arch; // nullable (B,T,H) archive for h2
};

__global__ void k_lstm_step2(StepArgs sa, StepArgs sm, int t) {
  const StepArgs s = (blockIdx.y == 0) ? sa : sm;

  const int lane = threadIdx.x & 31;
  const int hl   = lane >> 4;
  const int l    = lane & 15;
  const int wid  = blockIdx.x * (blockDim.x >> 5) + (threadIdx.x >> 5);
  const int mt   = wid & 1;    // 2 M tiles (B = 32)
  const int ng   = wid >> 1;   // 64 N groups over H

  const int arow = mt * 16 + l;
  const bf16_t* a1row = s.A1 + (size_t)arow * H;
  const bf16_t* w1row[4];
#pragma unroll
  for (int q = 0; q < 4; ++q)
    w1row[q] = s.W1 + (size_t)(q * H + ng * 16 + l) * H;

  v8f acc[4] = {v8f{}, v8f{}, v8f{}, v8f{}};
  for (int k = 0; k < H; k += 32) {
    v16bf a = load_a_bf(a1row, k, hl);
    __builtin_prefetch(w1row[0] + k + 256, 0, 1);
#pragma unroll
    for (int q = 0; q < 4; ++q)
      acc[q] = wmma_bf16(a, load_b_bf(w1row[q], k, hl), acc[q]);
  }

  if (s.A2) {
    const bf16_t* a2row = s.A2 + (size_t)arow * H;
    const bf16_t* w2row[4];
#pragma unroll
    for (int q = 0; q < 4; ++q)
      w2row[q] = s.W2 + (size_t)(q * H + ng * 16 + l) * H;
    for (int k = 0; k < H; k += 32) {
      v16bf a = load_a_bf(a2row, k, hl);
      __builtin_prefetch(w2row[0] + k + 256, 0, 1);
#pragma unroll
      for (int q = 0; q < 4; ++q)
        acc[q] = wmma_bf16(a, load_b_bf(w2row[q], k, hl), acc[q]);
    }
  }

  const int n = ng * 16 + l;
  float bias[4];
#pragma unroll
  for (int q = 0; q < 4; ++q)
    bias[q] = s.Gbase ? 0.0f : (s.bih[q * H + n] + s.bhh[q * H + n]);

#pragma unroll
  for (int r = 0; r < 8; ++r) {
    const int m = mt * 16 + r + hl * 8;   // batch index
    float gi = acc[0][r] + bias[0];
    float gf = acc[1][r] + bias[1];
    float gg = acc[2][r] + bias[2];
    float go = acc[3][r] + bias[3];
    if (s.Gbase) {
      const float* gb = s.Gbase + (size_t)m * H4 + n;
      gi += gb[0 * H]; gf += gb[1 * H]; gg += gb[2 * H]; go += gb[3 * H];
    }
    const size_t idx = (size_t)m * H + n;
    float cn = sigmoidf_(gf) * s.c[idx] + sigmoidf_(gi) * tanhf(gg);
    float hn = sigmoidf_(go) * tanhf(cn);
    s.c[idx]     = cn;
    s.h_out[idx] = (__bf16)hn;
    if (s.h_arch) s.h_arch[((size_t)m * T + t) * H + n] = (__bf16)hn;
  }
}

// ---------------------------------------------------------------------------
// Kernel: fused gated two-stream fusion.
// g = sigmoid(concat(Ha,Hm) @ Wg^T + bg)
// out = g * tanh(Ha @ Wa^T + ba) + (1-g) * tanh(Hm @ Wm^T + bm)
// 3 accumulator chains per wave; rows r = b*T + t match output layout.
// ---------------------------------------------------------------------------
__global__ void k_fusion(const bf16_t* __restrict__ Ha,  // (BT,H)
                         const bf16_t* __restrict__ Hm,  // (BT,H)
                         const bf16_t* __restrict__ Wg,  // (H,2H)
                         const bf16_t* __restrict__ Wa,  // (H,H)
                         const bf16_t* __restrict__ Wm,  // (H,H)
                         const float* __restrict__ bg,
                         const float* __restrict__ ba,
                         const float* __restrict__ bm,
                         float* __restrict__ out)        // (BT,H)
{
  const int lane = threadIdx.x & 31;
  const int hl   = lane >> 4;
  const int l    = lane & 15;
  const int wid  = blockIdx.x * (blockDim.x >> 5) + (threadIdx.x >> 5);
  const int nt   = wid & 63;    // 64 N tiles over H
  const int mt   = wid >> 6;    // 128 M tiles over BT

  const bf16_t* harow = Ha + (size_t)(mt * 16 + l) * H;
  const bf16_t* hmrow = Hm + (size_t)(mt * 16 + l) * H;
  const int n = nt * 16 + l;
  const bf16_t* wgrow = Wg + (size_t)n * H2;
  const bf16_t* warow = Wa + (size_t)n * H;
  const bf16_t* wmrow = Wm + (size_t)n * H;

  v8f accg = {}, acca = {}, accm = {};
  for (int k = 0; k < H; k += 32) {
    v16bf aa = load_a_bf(harow, k, hl);
    v16bf am = load_a_bf(hmrow, k, hl);
    __builtin_prefetch(wgrow + k + 256, 0, 1);
    accg = wmma_bf16(aa, load_b_bf(wgrow, k, hl), accg);       // Ha half of Wg
    accg = wmma_bf16(am, load_b_bf(wgrow, H + k, hl), accg);   // Hm half of Wg
    acca = wmma_bf16(aa, load_b_bf(warow, k, hl), acca);
    accm = wmma_bf16(am, load_b_bf(wmrow, k, hl), accm);
  }

  const float bgn = bg[n], ban = ba[n], bmn = bm[n];
#pragma unroll
  for (int r = 0; r < 8; ++r) {
    const int row = mt * 16 + r + hl * 8;
    float g = sigmoidf_(accg[r] + bgn);
    out[(size_t)row * H + n] =
        g * tanhf(acca[r] + ban) + (1.0f - g) * tanhf(accm[r] + bmn);
  }
}

// ---------------------------------------------------------------------------
// Workspace layout (bytes). Everything naturally aligned (powers of two).
// ---------------------------------------------------------------------------
constexpr size_t E_HST   = (size_t)B * H;      // one h/c state
constexpr size_t E_GX    = (size_t)BT * H4;    // precomputed input gates
constexpr size_t E_HARC  = (size_t)BT * H;     // archived h2 per stream
constexpr size_t E_X     = (size_t)BT * H2;    // one input stream
constexpr size_t E_W1IH  = (size_t)H4 * H2;    // cell-1 Wih
constexpr size_t E_WHH   = (size_t)H4 * H;     // 4HxH weights
constexpr size_t E_WG    = (size_t)H * H2;
constexpr size_t E_WHX   = (size_t)H * H;

constexpr size_t OFF_H      = 0;                              // 8 bf16 h slots
constexpr size_t OFF_C      = OFF_H      + 8 * E_HST * 2;     // 4 fp32 c states
constexpr size_t OFF_GXA    = OFF_C      + 4 * E_HST * 4;
constexpr size_t OFF_GXM    = OFF_GXA    + E_GX * 4;
constexpr size_t OFF_W1A_IH = OFF_GXM    + E_GX * 4;
constexpr size_t OFF_W1A_HH = OFF_W1A_IH + E_W1IH * 2;
constexpr size_t OFF_W2A_IH = OFF_W1A_HH + E_WHH * 2;
constexpr size_t OFF_W2A_HH = OFF_W2A_IH + E_WHH * 2;
constexpr size_t OFF_W1M_IH = OFF_W2A_HH + E_WHH * 2;
constexpr size_t OFF_W1M_HH = OFF_W1M_IH + E_W1IH * 2;
constexpr size_t OFF_W2M_IH = OFF_W1M_HH + E_WHH * 2;
constexpr size_t OFF_W2M_HH = OFF_W2M_IH + E_WHH * 2;
constexpr size_t OFF_WG     = OFF_W2M_HH + E_WHH * 2;
constexpr size_t OFF_WA     = OFF_WG     + E_WG * 2;
constexpr size_t OFF_WM     = OFF_WA     + E_WHX * 2;
constexpr size_t OFF_HAB    = OFF_WM     + E_WHX * 2;
constexpr size_t OFF_HMB    = OFF_HAB    + E_HARC * 2;
constexpr size_t OFF_XA     = OFF_HMB    + E_HARC * 2;
constexpr size_t OFF_XM     = OFF_XA     + E_X * 2;

// ---------------------------------------------------------------------------
extern "C" void kernel_launch(void* const* d_in, const int* in_sizes, int n_in,
                              void* d_out, int out_size, void* d_ws,
                              size_t ws_size, hipStream_t stream) {
  (void)in_sizes; (void)n_in; (void)out_size; (void)ws_size;

  const float* Xa    = (const float*)d_in[0];
  const float* Xm    = (const float*)d_in[1];
  const float* Wih1a = (const float*)d_in[2];
  const float* Whh1a = (const float*)d_in[3];
  const float* bih1a = (const float*)d_in[4];
  const float* bhh1a = (const float*)d_in[5];
  const float* Wih2a = (const float*)d_in[6];
  const float* Whh2a = (const float*)d_in[7];
  const float* bih2a = (const float*)d_in[8];
  const float* bhh2a = (const float*)d_in[9];
  const float* Wih1m = (const float*)d_in[10];
  const float* Whh1m = (const float*)d_in[11];
  const float* bih1m = (const float*)d_in[12];
  const float* bhh1m = (const float*)d_in[13];
  const float* Wih2m = (const float*)d_in[14];
  const float* Whh2m = (const float*)d_in[15];
  const float* bih2m = (const float*)d_in[16];
  const float* bhh2m = (const float*)d_in[17];
  const float* Wg    = (const float*)d_in[18];
  const float* bg    = (const float*)d_in[19];
  const float* Wa    = (const float*)d_in[20];
  const float* ba    = (const float*)d_in[21];
  const float* Wm    = (const float*)d_in[22];
  const float* bm    = (const float*)d_in[23];

  char* ws = (char*)d_ws;
  auto BF = [&](size_t off) { return (bf16_t*)(ws + off); };
  auto F  = [&](size_t off) { return (float*)(ws + off); };

  // h state slots: cell 0=h1a 1=h2a 2=h1m 3=h2m, each with 2 ping-pong slots
  auto hbf = [&](int cell, int slot) {
    return BF(OFF_H + (size_t)(cell * 2 + slot) * E_HST * 2);
  };
  auto cst = [&](int cell) { return F(OFF_C + (size_t)cell * E_HST * 4); };

  // --- 1) weights + inputs fp32 -> bf16 (one pass; L2-resident afterwards) ---
  auto cvt = [&](const float* s, bf16_t* d, size_t nelem) {
    int n4     = (int)(nelem / 4);
    int blocks = (n4 + 1023) / 1024;
    if (blocks > 4096) blocks = 4096;
    k_f32_to_bf16<<<blocks, 256, 0, stream>>>(s, d, n4);
  };
  cvt(Wih1a, BF(OFF_W1A_IH), E_W1IH);
  cvt(Whh1a, BF(OFF_W1A_HH), E_WHH);
  cvt(Wih2a, BF(OFF_W2A_IH), E_WHH);
  cvt(Whh2a, BF(OFF_W2A_HH), E_WHH);
  cvt(Wih1m, BF(OFF_W1M_IH), E_W1IH);
  cvt(Whh1m, BF(OFF_W1M_HH), E_WHH);
  cvt(Wih2m, BF(OFF_W2M_IH), E_WHH);
  cvt(Whh2m, BF(OFF_W2M_HH), E_WHH);
  cvt(Wg,    BF(OFF_WG),     E_WG);
  cvt(Wa,    BF(OFF_WA),     E_WHX);
  cvt(Wm,    BF(OFF_WM),     E_WHX);
  cvt(Xa,    BF(OFF_XA),     E_X);
  cvt(Xm,    BF(OFF_XM),     E_X);

  // --- 2) zero h/c states (1 MB contiguous region) ---
  {
    int n64 = (int)((8 * E_HST * 2 + 4 * E_HST * 4) / 8);
    k_zero_u64<<<(n64 + 255) / 256, 256, 0, stream>>>(
        (unsigned long long*)(ws + OFF_H), n64);
  }

  // --- 3) parallel precompute of cell-1 input projections (big WMMA GEMMs) ---
  // (T*B x 2H) @ (2H x 4H) per stream; 8192 waves, 4 N-tiles each.
  k_gemm_gx<<<1024, 256, 0, stream>>>(BF(OFF_XA), BF(OFF_W1A_IH), bih1a, bhh1a,
                                      F(OFF_GXA));
  k_gemm_gx<<<1024, 256, 0, stream>>>(BF(OFF_XM), BF(OFF_W1M_IH), bih1m, bhh1m,
                                      F(OFF_GXM));

  // --- 4) serial recurrence: 64 steps x 2 dual-stream launches ---
  int p = 0;
  for (int t = 0; t < T; ++t) {
    const float* gxa_t = F(OFF_GXA) + (size_t)t * B * H4;
    const float* gxm_t = F(OFF_GXM) + (size_t)t * B * H4;

    StepArgs c1a = {hbf(0, p), BF(OFF_W1A_HH), nullptr, nullptr,
                    gxa_t, nullptr, nullptr, cst(0), hbf(0, 1 - p), nullptr};
    StepArgs c1m = {hbf(2, p), BF(OFF_W1M_HH), nullptr, nullptr,
                    gxm_t, nullptr, nullptr, cst(2), hbf(2, 1 - p), nullptr};
    k_lstm_step2<<<dim3(16, 2), 256, 0, stream>>>(c1a, c1m, t);

    StepArgs c2a = {hbf(0, 1 - p), BF(OFF_W2A_IH), hbf(1, p), BF(OFF_W2A_HH),
                    nullptr, bih2a, bhh2a, cst(1), hbf(1, 1 - p), BF(OFF_HAB)};
    StepArgs c2m = {hbf(2, 1 - p), BF(OFF_W2M_IH), hbf(3, p), BF(OFF_W2M_HH),
                    nullptr, bih2m, bhh2m, cst(3), hbf(3, 1 - p), BF(OFF_HMB)};
    k_lstm_step2<<<dim3(16, 2), 256, 0, stream>>>(c2a, c2m, t);

    p ^= 1;
  }

  // --- 5) fused gated two-stream fusion over all (B*T, H) ---
  k_fusion<<<1024, 256, 0, stream>>>(BF(OFF_HAB), BF(OFF_HMB), BF(OFF_WG),
                                     BF(OFF_WA), BF(OFF_WM), bg, ba, bm,
                                     (float*)d_out);
}
